// GraphModule_37538014167219
// MI455X (gfx1250) — compile-verified, matrix operand order checked by reading, not run on hardware
//
#include <hip/hip_runtime.h>

#define USE_ASYNC_LDS 1   // CDNA5 global_load_async_to_lds_b128 staging path

// ---------------------------------------------------------------------------
// Types for CDNA5 WMMA (wave32, 16x16x32 bf16 -> f32)
// ---------------------------------------------------------------------------
typedef __bf16 bf16;
typedef __attribute__((ext_vector_type(16))) __bf16 v16bf;
typedef __attribute__((ext_vector_type(8)))  __bf16 v8bf;
typedef __attribute__((ext_vector_type(8)))  float  v8f;

// Problem constants
#define BW_   1024      // B*NW windows
#define NTOK  64        // tokens per window
#define HH    16        // heads
#define DHD   32        // head dim
#define CC    512
#define DFF_  2048
#define MTOT  65536     // BW_*NTOK
#define NCLS_ 1000
#define NB_   16        // batch

__device__ __forceinline__ v16bf ldfrag(const bf16* p) {
    // elements 0..7 at p[0..7], 8..15 at p[16..23] (ISA 16-bit operand layout)
    v8bf lo = *(const v8bf*)p;
    v8bf hi = *(const v8bf*)(p + 16);
    return __builtin_shufflevector(lo, hi, 0,1,2,3,4,5,6,7,8,9,10,11,12,13,14,15);
}

__device__ __forceinline__ v8f wmma_bf16(v16bf a, v16bf b, v8f c) {
    return __builtin_amdgcn_wmma_f32_16x16x32_bf16(false, a, false, b, (short)0, c, false, false);
}

__device__ __forceinline__ void async_cp_b128(const bf16* gsrc, bf16* ldst) {
#if USE_ASYNC_LDS
    unsigned loff = (unsigned)(unsigned long long)ldst;   // low 32 bits = LDS byte offset
    asm volatile("global_load_async_to_lds_b128 %0, %1, off"
                 :: "v"(loff), "v"(gsrc) : "memory");
#else
    *(uint4*)ldst = *(const uint4*)gsrc;
#endif
}

__device__ __forceinline__ void wait_async_lds() {
#if USE_ASYNC_LDS
    asm volatile("s_wait_asynccnt 0" ::: "memory");
#endif
}

// ---------------------------------------------------------------------------
// fp32 -> bf16 convert (4 elems / thread)
// ---------------------------------------------------------------------------
__global__ __launch_bounds__(256) void f2bf_kernel(const float* __restrict__ x,
                                                   bf16* __restrict__ y, long n) {
    long i = ((long)blockIdx.x * 256 + threadIdx.x) * 4;
    if (i >= n) return;
    float4 v = *(const float4*)(x + i);
    y[i + 0] = (bf16)v.x; y[i + 1] = (bf16)v.y;
    y[i + 2] = (bf16)v.z; y[i + 3] = (bf16)v.w;
}

// ---------------------------------------------------------------------------
// L2-normalize 32-vectors (q/k) and convert to bf16
// ---------------------------------------------------------------------------
__global__ __launch_bounds__(256) void l2n_kernel(const float* __restrict__ x,
                                                  bf16* __restrict__ y, int nvec) {
    int i = blockIdx.x * 256 + threadIdx.x;
    if (i >= nvec) return;
    const float* p = x + (long)i * 32;
    float s = 0.f;
#pragma unroll
    for (int k = 0; k < 32; ++k) s += p[k] * p[k];
    float inv = 1.f / fmaxf(sqrtf(s), 1e-12f);
    bf16* q = y + (long)i * 32;
#pragma unroll
    for (int k = 0; k < 32; ++k) q[k] = (bf16)(p[k] * inv);
}

// ---------------------------------------------------------------------------
// CPB MLP: rel[225][16] = relu(t @ w0^T + b0) @ w2^T
// ---------------------------------------------------------------------------
__global__ __launch_bounds__(256) void cpb_kernel(const float* __restrict__ ct,
                                                  const float* __restrict__ w0,
                                                  const float* __restrict__ b0,
                                                  const float* __restrict__ w2,
                                                  float* __restrict__ rel) {
    __shared__ float red[256];
    int i = blockIdx.x;                       // 0..224
    float t0 = ct[i * 2 + 0], t1 = ct[i * 2 + 1];
    float acc[16];
#pragma unroll
    for (int h = 0; h < 16; ++h) acc[h] = 0.f;
    for (int j = threadIdx.x; j < 512; j += 256) {
        float hj = fmaxf(t0 * w0[j * 2] + t1 * w0[j * 2 + 1] + b0[j], 0.f);
#pragma unroll
        for (int h = 0; h < 16; ++h) acc[h] += hj * w2[h * 512 + j];
    }
    for (int h = 0; h < 16; ++h) {
        red[threadIdx.x] = acc[h]; __syncthreads();
        for (int st = 128; st > 0; st >>= 1) {
            if (threadIdx.x < st) red[threadIdx.x] += red[threadIdx.x + st];
            __syncthreads();
        }
        if (threadIdx.x == 0) rel[i * 16 + h] = red[0];
        __syncthreads();
    }
}

// bias16[h][r][c] = 16*sigmoid(rel[rpi[r*64+c]][h])
__global__ __launch_bounds__(256) void bias_kernel(const float* __restrict__ rel,
                                                   const int* __restrict__ rpi,
                                                   float* __restrict__ bias16) {
    int idx = blockIdx.x * 256 + threadIdx.x;
    if (idx >= 16 * 4096) return;
    int h = idx >> 12, rc = idx & 4095;
    float x = rel[rpi[rc] * 16 + h];
    bias16[idx] = 16.f / (1.f + __expf(-x));
}

// ---------------------------------------------------------------------------
// Tiled bf16 NT GEMM: out = A[M,K] @ W[N,K]^T + bias
// BM=128 BN=128 BK=64; 256 thr = 8 waves (2x4); wave tile 64x32.
// Staging: async global->LDS (ASYNCcnt), double buffered.
// EPI: 0=f32 store, 1=bf16 store, 2=gelu+bf16, 3=vT transposed bf16 store
// ---------------------------------------------------------------------------
template <int EPI>
__global__ __launch_bounds__(256) void gemm_bf16_nt(const bf16* __restrict__ A,
                                                    const bf16* __restrict__ W,
                                                    const float* __restrict__ bias,
                                                    void* __restrict__ out,
                                                    int M, int N, int K) {
    constexpr int LDT = 72;                       // padded LDS row (bank-conflict free)
    __shared__ bf16 As[2][128][LDT];
    __shared__ bf16 Bs[2][128][LDT];

    const int tid  = threadIdx.x;
    const int lane = tid & 31;
    const int wid  = tid >> 5;
    const int wm   = (wid >> 2) * 64;             // wave M origin (2 rows of waves)
    const int wn   = (wid & 3) * 32;              // wave N origin (4 cols of waves)
    const int bm0  = blockIdx.y * 128;
    const int bn0  = blockIdx.x * 128;
    const int lr   = tid >> 1;                    // staging row 0..127
    const int lc   = (tid & 1) * 32;              // staging col 0 or 32

    v8f acc[4][2] = {};

    auto stage = [&](int s, int k0) {
        const bf16* ga = A + (long)(bm0 + lr) * K + k0 + lc;
        const bf16* gb = W + (long)(bn0 + lr) * K + k0 + lc;
        bf16* la = &As[s][lr][lc];
        bf16* lb = &Bs[s][lr][lc];
#pragma unroll
        for (int q = 0; q < 4; ++q) {
            async_cp_b128(ga + q * 8, la + q * 8);
            async_cp_b128(gb + q * 8, lb + q * 8);
        }
    };

    stage(0, 0);
    wait_async_lds();
    __syncthreads();

    const int nk = K >> 6;
    const int kh = (lane >> 4) * 8;
    const int rr = lane & 15;

    for (int kb = 0; kb < nk; ++kb) {
        int cur = kb & 1;
        if (kb + 1 < nk) stage(cur ^ 1, (kb + 1) * 64);
#pragma unroll
        for (int ks = 0; ks < 2; ++ks) {
            int k0 = ks * 32;
            v16bf af[4], bfr[2];
#pragma unroll
            for (int i = 0; i < 4; ++i) af[i]  = ldfrag(&As[cur][wm + i * 16 + rr][k0 + kh]);
#pragma unroll
            for (int j = 0; j < 2; ++j) bfr[j] = ldfrag(&Bs[cur][wn + j * 16 + rr][k0 + kh]);
#pragma unroll
            for (int i = 0; i < 4; ++i)
#pragma unroll
                for (int j = 0; j < 2; ++j) acc[i][j] = wmma_bf16(af[i], bfr[j], acc[i][j]);
        }
        wait_async_lds();
        __syncthreads();
    }

    // Epilogue. C layout: element e -> row = e + 8*(lane>>4), col = lane&15
    const int half = lane >> 4;
#pragma unroll
    for (int i = 0; i < 4; ++i) {
#pragma unroll
        for (int j = 0; j < 2; ++j) {
            int ng = bn0 + wn + j * 16 + rr;
            float bv = bias[ng];
            int mg = bm0 + wm + i * 16 + 8 * half;
#pragma unroll
            for (int e = 0; e < 8; ++e) {
                float v = acc[i][j][e] + bv;
                int row = mg + e;
                if (EPI == 0) {
                    ((float*)out)[(long)row * N + ng] = v;
                } else if (EPI == 1) {
                    ((bf16*)out)[(long)row * N + ng] = (bf16)v;
                } else if (EPI == 2) {
                    float g = 0.5f * v * (1.f + erff(v * 0.70710678118f));
                    ((bf16*)out)[(long)row * N + ng] = (bf16)g;
                } else { // EPI == 3 : vT[bw][h][dh][tok]
                    long o = (((long)(row >> 6) * HH + (ng >> 5)) * DHD + (ng & 31)) * NTOK + (row & 63);
                    ((bf16*)out)[o] = (bf16)v;
                }
            }
        }
    }
}

// ---------------------------------------------------------------------------
// Attention scores + fused softmax.  One block = one (bw,h); 4 waves.
// attn = (qn @ kn^T) * scale + bias ; softmax rows ; bf16 probs out.
// ---------------------------------------------------------------------------
__global__ __launch_bounds__(128) void attn_scores_kernel(const bf16* __restrict__ qn,
                                                          const bf16* __restrict__ kn,
                                                          const float* __restrict__ bias16,
                                                          const float* __restrict__ logit_scale,
                                                          bf16* __restrict__ P) {
    int bh = blockIdx.x;                 // bw*16 + h
    int h  = bh & 15;
    int lane = threadIdx.x & 31;
    int wid  = threadIdx.x >> 5;
    int m0   = wid * 16;
    int rr   = lane & 15;
    int kh   = (lane >> 4) * 8;
    int half = lane >> 4;

    const bf16* qb = qn + (long)bh * (NTOK * DHD);
    const bf16* kb = kn + (long)bh * (NTOK * DHD);

    v16bf a = ldfrag(qb + (m0 + rr) * DHD + kh);
    v8f acc[4] = {};
#pragma unroll
    for (int j = 0; j < 4; ++j) {
        v16bf b = ldfrag(kb + (j * 16 + rr) * DHD + kh);
        acc[j] = wmma_bf16(a, b, acc[j]);
    }

    float scale = __expf(fminf(logit_scale[h], 4.6051702f));   // min(ls, log(100))
    const float* bh_bias = bias16 + (long)h * 4096;

    float s[4][8];
#pragma unroll
    for (int j = 0; j < 4; ++j) {
        int col = j * 16 + rr;
#pragma unroll
        for (int e = 0; e < 8; ++e) {
            int row = m0 + 8 * half + e;
            s[j][e] = acc[j][e] * scale + bh_bias[row * 64 + col];
        }
    }
    // row softmax: row fixed by e (+half); 16 columns/tile spread over the 16-lane half
#pragma unroll
    for (int e = 0; e < 8; ++e) {
        float mx = fmaxf(fmaxf(s[0][e], s[1][e]), fmaxf(s[2][e], s[3][e]));
#pragma unroll
        for (int d = 1; d <= 8; d <<= 1) mx = fmaxf(mx, __shfl_xor(mx, d, 32));
        float sum = 0.f;
#pragma unroll
        for (int j = 0; j < 4; ++j) { s[j][e] = __expf(s[j][e] - mx); sum += s[j][e]; }
#pragma unroll
        for (int d = 1; d <= 8; d <<= 1) sum += __shfl_xor(sum, d, 32);
        float inv = 1.f / sum;
        int row = m0 + 8 * half + e;
        bf16* prow = P + (long)bh * 4096 + row * 64;
#pragma unroll
        for (int j = 0; j < 4; ++j) prow[j * 16 + rr] = (bf16)(s[j][e] * inv);
    }
}

// ---------------------------------------------------------------------------
// Batched ctx = P @ v : per (bw,h), [64x64]@[64x32] using vT[bw][h][dh][tok].
// ctx stored bf16 row-major [MTOT, C] at channel h*32+dh.
// ---------------------------------------------------------------------------
__global__ __launch_bounds__(128) void attn_ctx_kernel(const bf16* __restrict__ P,
                                                       const bf16* __restrict__ vT,
                                                       bf16* __restrict__ ctx) {
    int bh = blockIdx.x;
    int bw = bh >> 4, h = bh & 15;
    int lane = threadIdx.x & 31;
    int wid  = threadIdx.x >> 5;
    int m0   = wid * 16;
    int rr   = lane & 15;
    int kh   = (lane >> 4) * 8;
    int half = lane >> 4;

    const bf16* pb = P  + (long)bh * 4096;   // [64 tok][64 tok]
    const bf16* vb = vT + (long)bh * 2048;   // [32 dh][64 tok]

    v8f acc[2] = {};
#pragma unroll
    for (int ks = 0; ks < 2; ++ks) {
        int k0 = ks * 32;
        v16bf a = ldfrag(pb + (m0 + rr) * 64 + k0 + kh);
#pragma unroll
        for (int j = 0; j < 2; ++j) {
            v16bf b = ldfrag(vb + (j * 16 + rr) * 64 + k0 + kh);
            acc[j] = wmma_bf16(a, b, acc[j]);
        }
    }
#pragma unroll
    for (int j = 0; j < 2; ++j) {
        int dh = j * 16 + rr;
#pragma unroll
        for (int e = 0; e < 8; ++e) {
            int tok = m0 + 8 * half + e;
            ctx[((long)bw * NTOK + tok) * CC + h * DHD + dh] = (bf16)acc[j][e];
        }
    }
}

// ---------------------------------------------------------------------------
// hidden = input_feature + LN(attn_out); write f32 + bf16
// ---------------------------------------------------------------------------
__global__ __launch_bounds__(256) void ln1_kernel(const float* __restrict__ ao,
                                                  const float* __restrict__ inpf,
                                                  const float* __restrict__ w,
                                                  const float* __restrict__ b,
                                                  float* __restrict__ hf,
                                                  bf16* __restrict__ hb) {
    __shared__ float red[256];
    long row = blockIdx.x;
    const float* x = ao + row * CC;
    int t = threadIdx.x;
    float a0 = x[t], a1 = x[t + 256];

    red[t] = a0 + a1; __syncthreads();
    for (int st = 128; st > 0; st >>= 1) { if (t < st) red[t] += red[t + st]; __syncthreads(); }
    float mean = red[0] * (1.f / CC); __syncthreads();
    float d0 = a0 - mean, d1 = a1 - mean;
    red[t] = d0 * d0 + d1 * d1; __syncthreads();
    for (int st = 128; st > 0; st >>= 1) { if (t < st) red[t] += red[t + st]; __syncthreads(); }
    float rstd = rsqrtf(red[0] * (1.f / CC) + 1e-5f);

    float h0 = inpf[row * CC + t]       + d0 * rstd * w[t]       + b[t];
    float h1 = inpf[row * CC + t + 256] + d1 * rstd * w[t + 256] + b[t + 256];
    hf[row * CC + t] = h0;       hf[row * CC + t + 256] = h1;
    hb[row * CC + t] = (bf16)h0; hb[row * CC + t + 256] = (bf16)h1;
}

// ---------------------------------------------------------------------------
// layer_out = hidden + LN(mlp); final = LN(layer_out)
// ---------------------------------------------------------------------------
__global__ __launch_bounds__(256) void ln2_kernel(const float* __restrict__ hf,
                                                  const float* __restrict__ mlp,
                                                  const float* __restrict__ law, const float* __restrict__ lab,
                                                  const float* __restrict__ flw, const float* __restrict__ flb,
                                                  float* __restrict__ fout) {
    __shared__ float red[256];
    long row = blockIdx.x;
    int t = threadIdx.x;
    const float* x = mlp + row * CC;
    float a0 = x[t], a1 = x[t + 256];

    red[t] = a0 + a1; __syncthreads();
    for (int st = 128; st > 0; st >>= 1) { if (t < st) red[t] += red[t + st]; __syncthreads(); }
    float mean = red[0] * (1.f / CC); __syncthreads();
    float d0 = a0 - mean, d1 = a1 - mean;
    red[t] = d0 * d0 + d1 * d1; __syncthreads();
    for (int st = 128; st > 0; st >>= 1) { if (t < st) red[t] += red[t + st]; __syncthreads(); }
    float rstd = rsqrtf(red[0] * (1.f / CC) + 1e-5f); __syncthreads();

    float l0 = hf[row * CC + t]       + d0 * rstd * law[t]       + lab[t];
    float l1 = hf[row * CC + t + 256] + d1 * rstd * law[t + 256] + lab[t + 256];

    red[t] = l0 + l1; __syncthreads();
    for (int st = 128; st > 0; st >>= 1) { if (t < st) red[t] += red[t + st]; __syncthreads(); }
    float m2 = red[0] * (1.f / CC); __syncthreads();
    float e0 = l0 - m2, e1 = l1 - m2;
    red[t] = e0 * e0 + e1 * e1; __syncthreads();
    for (int st = 128; st > 0; st >>= 1) { if (t < st) red[t] += red[t + st]; __syncthreads(); }
    float r2 = rsqrtf(red[0] * (1.f / CC) + 1e-5f);

    fout[row * CC + t]       = e0 * r2 * flw[t]       + flb[t];
    fout[row * CC + t + 256] = e1 * r2 * flw[t + 256] + flb[t + 256];
}

// mean pool over 4096 tokens per batch
__global__ __launch_bounds__(256) void pool_kernel(const float* __restrict__ F,
                                                   float* __restrict__ pooled) {
    int t = blockIdx.x * 256 + threadIdx.x;       // 8192 = 16*512
    if (t >= NB_ * CC) return;
    int b = t >> 9, c = t & 511;
    const float* p = F + ((long)b * 4096) * CC + c;
    float s = 0.f;
    for (int i = 0; i < 4096; ++i) s += p[(long)i * CC];
    pooled[t] = s * (1.f / 4096.f);
}

__global__ __launch_bounds__(256) void cls_kernel(const float* __restrict__ pooled,
                                                  const float* __restrict__ cw,
                                                  const float* __restrict__ cb,
                                                  float* __restrict__ out) {
    int t = blockIdx.x * 256 + threadIdx.x;
    if (t >= NB_ * NCLS_) return;
    int b = t / NCLS_, k = t % NCLS_;
    const float* pp = pooled + b * CC;
    const float* pw = cw + (long)k * CC;
    float s = cb[k];
    for (int i = 0; i < CC; ++i) s += pp[i] * pw[i];
    out[t] = s;
}

// ---------------------------------------------------------------------------
// launch
// ---------------------------------------------------------------------------
extern "C" void kernel_launch(void* const* d_in, const int* in_sizes, int n_in,
                              void* d_out, int out_size, void* d_ws, size_t ws_size,
                              hipStream_t stream) {
    const float* cls_w    = (const float*)d_in[0];
    const float* cls_b    = (const float*)d_in[1];
    const float* v_w      = (const float*)d_in[2];
    const float* v_b      = (const float*)d_in[3];
    const float* aow      = (const float*)d_in[4];
    const float* aob      = (const float*)d_in[5];
    const float* lscale   = (const float*)d_in[6];
    const float* ct       = (const float*)d_in[7];
    const float* cpb_w0   = (const float*)d_in[8];
    const float* cpb_b0   = (const float*)d_in[9];
    const float* cpb_w2   = (const float*)d_in[10];
    const float* lbw      = (const float*)d_in[11];
    const float* lbb      = (const float*)d_in[12];
    const float* inter_w  = (const float*)d_in[13];
    const float* inter_b  = (const float*)d_in[14];
    const float* out_w    = (const float*)d_in[15];
    const float* out_b    = (const float*)d_in[16];
    const float* law      = (const float*)d_in[17];
    const float* lab      = (const float*)d_in[18];
    const float* flw      = (const float*)d_in[19];
    const float* flb      = (const float*)d_in[20];
    const float* hsw      = (const float*)d_in[21];
    const float* inpf     = (const float*)d_in[22];
    const float* qlay     = (const float*)d_in[23];
    const float* klay     = (const float*)d_in[24];
    const int*   rpi      = (const int*)  d_in[25];

    char* ws = (char*)d_ws;
    const size_t MB = 1ull << 20;
    // small regions
    bf16*  VW   = (bf16*)(ws + 0 * MB);            // 0.5 MB
    bf16*  AOW  = (bf16*)(ws + 1 * MB);            // 0.5 MB
    bf16*  IW   = (bf16*)(ws + 2 * MB);            // 2 MB
    bf16*  OW   = (bf16*)(ws + 4 * MB);            // 2 MB
    float* REL  = (float*)(ws + 6 * MB);           // 14.4 KB
    float* BIAS = (float*)(ws + 7 * MB);           // 256 KB
    float* POOL = (float*)(ws + 7 * MB + 512 * 1024);
    // big regions (with aliasing)
    const size_t BASE = 8 * MB;
    bf16*  XBF = (bf16*)(ws + BASE);               // 64 MB  (later: CTX)
    bf16*  QN  = (bf16*)(ws + BASE + 64 * MB);     // 64 MB  } later AOUT (f32,128MB), then F
    bf16*  KN  = (bf16*)(ws + BASE + 128 * MB);    // 64 MB  }
    bf16*  VT  = (bf16*)(ws + BASE + 192 * MB);    // 64 MB  (later: HB)
    bf16*  P   = (bf16*)(ws + BASE + 256 * MB);    // 128 MB (later: HF f32)
    bf16*  I   = (bf16*)(ws + BASE + 384 * MB);    // 256 MB
    float* MLP = (float*)(ws + BASE + 640 * MB);   // 128 MB
    bf16*  CTX = XBF;
    float* AOUT = (float*)QN;
    float* HF   = (float*)P;
    bf16*  HB   = VT;
    float* FOUT = (float*)QN;

    // 1. conversions
    f2bf_kernel<<<(512 * 512 / 4 + 255) / 256, 256, 0, stream>>>(v_w, VW, 512 * 512);
    f2bf_kernel<<<(512 * 512 / 4 + 255) / 256, 256, 0, stream>>>(aow, AOW, 512 * 512);
    f2bf_kernel<<<(2048 * 512 / 4 + 255) / 256, 256, 0, stream>>>(inter_w, IW, 2048 * 512);
    f2bf_kernel<<<(512 * 2048 / 4 + 255) / 256, 256, 0, stream>>>(out_w, OW, 512 * 2048);
    f2bf_kernel<<<((long)MTOT * CC / 4 + 255) / 256, 256, 0, stream>>>(hsw, XBF, (long)MTOT * CC);
    l2n_kernel<<<(BW_ * HH * NTOK + 255) / 256, 256, 0, stream>>>(qlay, QN, BW_ * HH * NTOK);
    l2n_kernel<<<(BW_ * HH * NTOK + 255) / 256, 256, 0, stream>>>(klay, KN, BW_ * HH * NTOK);

    // 2. relative position bias
    cpb_kernel<<<225, 256, 0, stream>>>(ct, cpb_w0, cpb_b0, cpb_w2, REL);
    bias_kernel<<<(16 * 4096 + 255) / 256, 256, 0, stream>>>(REL, rpi, BIAS);

    // 3. value projection -> vT[bw][h][dh][tok]
    gemm_bf16_nt<3><<<dim3(CC / 128, MTOT / 128), 256, 0, stream>>>(XBF, VW, v_b, VT, MTOT, CC, CC);

    // 4. attention
    attn_scores_kernel<<<BW_ * HH, 128, 0, stream>>>(QN, KN, BIAS, lscale, P);
    attn_ctx_kernel<<<BW_ * HH, 128, 0, stream>>>(P, VT, CTX);

    // 5. attention output projection (fp32 out)
    gemm_bf16_nt<0><<<dim3(CC / 128, MTOT / 128), 256, 0, stream>>>(CTX, AOW, aob, AOUT, MTOT, CC, CC);

    // 6. res-post-norm
    ln1_kernel<<<MTOT, 256, 0, stream>>>(AOUT, inpf, lbw, lbb, HF, HB);

    // 7. MLP
    gemm_bf16_nt<2><<<dim3(DFF_ / 128, MTOT / 128), 256, 0, stream>>>(HB, IW, inter_b, I, MTOT, DFF_, CC);
    gemm_bf16_nt<0><<<dim3(CC / 128, MTOT / 128), 256, 0, stream>>>(I, OW, out_b, MLP, MTOT, CC, DFF_);

    // 8. second LN + final LN
    ln2_kernel<<<MTOT, 256, 0, stream>>>(HF, MLP, law, lab, flw, flb, FOUT);

    // 9. pool + classifier
    pool_kernel<<<(NB_ * CC + 255) / 256, 256, 0, stream>>>(FOUT, POOL);
    cls_kernel<<<(NB_ * NCLS_ + 255) / 256, 256, 0, stream>>>(POOL, cls_w, cls_b, (float*)d_out);
}
